// MultiHeadAttentionPooling_22325240005209
// MI455X (gfx1250) — matrix operand order repaired and checked by reading, not compiled, not run on hardware
//
#include <hip/hip_runtime.h>
#include <math.h>

#define HIDDEN   512
#define HEADS    8
#define HALF_H   256           // HIDDEN/2
#define NROWS    262144
#define NGRAPH   1024
#define LN_EPS   1e-5f
#define EPS_W    1e-8f

// d_out layout: graph_z [G*HIDDEN] | weights [N*HEADS] | mean_entropy [1]
#define W_OFF    (NGRAPH * HIDDEN)            // 524288
#define ENT_OFF  (W_OFF + NROWS * HEADS)      // 2621440

typedef __attribute__((ext_vector_type(16))) _Float16 v16h;
typedef __attribute__((ext_vector_type(8)))  _Float16 v8h;
typedef __attribute__((ext_vector_type(8)))  float    v8f;
typedef __attribute__((ext_vector_type(4)))  float    v4f;

// ---------------------------------------------------------------------------
// Kernel 0: pack W1 -> f16 column-major [256][512], W2 -> f16 col-major
// padded [16][256] (cols 8..15 zero). Also zero the entropy accumulator.
// ---------------------------------------------------------------------------
__global__ void pack_weights(const float* __restrict__ W1,
                             const float* __restrict__ W2,
                             _Float16* __restrict__ w1cm,
                             _Float16* __restrict__ w2cm,
                             float* __restrict__ out) {
  if (blockIdx.x == 0 && threadIdx.x == 0) out[ENT_OFF] = 0.0f;
  const int T1 = HIDDEN * HALF_H;   // 131072  W1 elements
  const int T2 = 16 * HALF_H;       // 4096    padded W2 elements
  for (int idx = blockIdx.x * blockDim.x + threadIdx.x; idx < T1 + T2;
       idx += gridDim.x * blockDim.x) {
    if (idx < T1) {
      int k = idx / HALF_H;                 // 0..511
      int n = idx % HALF_H;                 // 0..255
      w1cm[n * HIDDEN + k] = (_Float16)W1[idx];
    } else {
      int j = idx - T1;
      int n = j / HALF_H;                   // 0..15
      int k = j % HALF_H;                   // 0..255
      float v = (n < HEADS) ? W2[k * HEADS + n] : 0.0f;
      w2cm[n * HALF_H + k] = (_Float16)v;
    }
  }
}

// ---------------------------------------------------------------------------
// Kernel 1: fused LayerNorm + MLP (WMMA f16) -> per-row logits [N, HEADS]
// 2 waves per block, each wave owns a 16-row tile.
// ---------------------------------------------------------------------------
__global__ __launch_bounds__(64)
void ln_mlp_logits(const float* __restrict__ x,
                   const float* __restrict__ gamma,
                   const float* __restrict__ beta,
                   const _Float16* __restrict__ w1cm,
                   const float* __restrict__ b1,
                   const _Float16* __restrict__ w2cm,
                   const float* __restrict__ b2,
                   float* __restrict__ logits) {
  __shared__ __attribute__((aligned(16))) _Float16 xn[2][16][HIDDEN];   // 32 KB
  __shared__ __attribute__((aligned(16))) _Float16 hb[2][16][HALF_H];   // 16 KB

  const int wave = threadIdx.x >> 5;
  const int lane = threadIdx.x & 31;
  const int row0 = blockIdx.x * 32 + wave * 16;
  const int lr   = lane & 15;      // row / column within fragment
  const int lh   = lane >> 4;      // half-wave selector

  // ---- LayerNorm: one row at a time, 32 lanes cooperate (coalesced) ----
  for (int rr = 0; rr < 16; ++rr) {
    const float* xr = x + (size_t)(row0 + rr) * HIDDEN + lane * 16;
    float vals[16];
#pragma unroll
    for (int i = 0; i < 16; i += 4) {
      v4f t = *(const v4f*)(xr + i);
      vals[i+0] = t[0]; vals[i+1] = t[1]; vals[i+2] = t[2]; vals[i+3] = t[3];
    }
    float s = 0.0f, q = 0.0f;
#pragma unroll
    for (int i = 0; i < 16; ++i) { s += vals[i]; q += vals[i] * vals[i]; }
#pragma unroll
    for (int off = 16; off > 0; off >>= 1) {
      s += __shfl_xor(s, off, 32);
      q += __shfl_xor(q, off, 32);
    }
    float mean = s * (1.0f / HIDDEN);
    float var  = q * (1.0f / HIDDEN) - mean * mean;
    float rstd = rsqrtf(var + LN_EPS);
    const float* gp = gamma + lane * 16;
    const float* bp = beta  + lane * 16;
    v8h o0, o1;
#pragma unroll
    for (int i = 0; i < 8; ++i)
      o0[i] = (_Float16)((vals[i] - mean) * rstd * gp[i] + bp[i]);
#pragma unroll
    for (int i = 0; i < 8; ++i)
      o1[i] = (_Float16)((vals[i+8] - mean) * rstd * gp[i+8] + bp[i+8]);
    *(v8h*)&xn[wave][rr][lane * 16 + 0] = o0;
    *(v8h*)&xn[wave][rr][lane * 16 + 8] = o1;
  }
  __syncthreads();

  // ---- GEMM1: x_norm[16x512] @ W1[512x256] via wmma_f32_16x16x32_f16 ----
  for (int nt = 0; nt < 16; ++nt) {
    v8f acc = {0.f, 0.f, 0.f, 0.f, 0.f, 0.f, 0.f, 0.f};
    const _Float16* bbase = w1cm + (size_t)(nt * 16 + lr) * HIDDEN + lh * 8;
    if (nt + 1 < 16)
      __builtin_prefetch(w1cm + (size_t)((nt + 1) * 16 + lr) * HIDDEN, 0, 0);
#pragma unroll
    for (int kt = 0; kt < 16; ++kt) {
      const int kb = kt * 32 + lh * 8;
      v8h a0 = *(const v8h*)&xn[wave][lr][kb];
      v8h a1 = *(const v8h*)&xn[wave][lr][kb + 16];
      const _Float16* bp = bbase + kt * 32;
      v8h b0 = *(const v8h*)(bp);
      v8h b1v = *(const v8h*)(bp + 16);
      v16h a, b;
#pragma unroll
      for (int i = 0; i < 8; ++i) {
        a[i] = a0[i]; a[i + 8] = a1[i];
        b[i] = b0[i]; b[i + 8] = b1v[i];
      }
      acc = __builtin_amdgcn_wmma_f32_16x16x32_f16(
          false, a, false, b, (short)0, acc, false, false);
    }
    // epilogue: bias + SiLU, stage h as f16 for GEMM2's A operand
    const int n = nt * 16 + lr;
    const float bias = b1[n];
#pragma unroll
    for (int j = 0; j < 8; ++j) {
      const int m = j + lh * 8;
      float v = acc[j] + bias;
      float sv = v / (1.0f + __expf(-v));
      hb[wave][m][n] = (_Float16)sv;
    }
  }
  __syncthreads();

  // ---- GEMM2: h[16x256] @ W2pad[256x16] -> logits tile 16x16 (8 valid) ----
  v8f acc2 = {0.f, 0.f, 0.f, 0.f, 0.f, 0.f, 0.f, 0.f};
#pragma unroll
  for (int kt = 0; kt < 8; ++kt) {
    const int kb = kt * 32 + lh * 8;
    v8h a0 = *(const v8h*)&hb[wave][lr][kb];
    v8h a1 = *(const v8h*)&hb[wave][lr][kb + 16];
    const _Float16* bp = w2cm + lr * HALF_H + kt * 32 + lh * 8;
    v8h b0 = *(const v8h*)(bp);
    v8h b1v = *(const v8h*)(bp + 16);
    v16h a, b;
#pragma unroll
    for (int i = 0; i < 8; ++i) {
      a[i] = a0[i]; a[i + 8] = a1[i];
      b[i] = b0[i]; b[i + 8] = b1v[i];
    }
    acc2 = __builtin_amdgcn_wmma_f32_16x16x32_f16(
        false, a, false, b, (short)0, acc2, false, false);
  }
  if (lr < HEADS) {
    const float bias = b2[lr];
#pragma unroll
    for (int j = 0; j < 8; ++j) {
      const int m = j + lh * 8;
      logits[(size_t)(row0 + m) * HEADS + lr] = acc2[j] + bias;
    }
  }
}

// ---------------------------------------------------------------------------
// Kernel 2: per-graph segment softmax (8 heads), entropy, weighted pooling.
// One block per graph; batch ids are sorted so segments are contiguous.
// Logits live in the weights region of d_out and are normalized in place.
// ---------------------------------------------------------------------------
__global__ __launch_bounds__(256)
void segment_softmax_pool(const float* __restrict__ x,
                          const int* __restrict__ batch,
                          float* __restrict__ out) {
  float* gz  = out;            // [G, HIDDEN]
  float* wts = out + W_OFF;    // logits in, weights out  [N, HEADS]
  const int g   = blockIdx.x;
  const int tid = threadIdx.x;
  const int wid = tid >> 5, ln = tid & 31;

  // binary search segment bounds
  int lo = 0, hi = NROWS;
  while (lo < hi) { int mid = (lo + hi) >> 1; if (batch[mid] < g) lo = mid + 1; else hi = mid; }
  const int start = lo;
  hi = NROWS;
  while (lo < hi) { int mid = (lo + hi) >> 1; if (batch[mid] < g + 1) lo = mid + 1; else hi = mid; }
  const int end = lo;

  __shared__ float warpred[8][8];
  __shared__ float smax[HEADS], sinv[HEADS];

  // ---- pass 1: per-head max ----
  float m[HEADS];
#pragma unroll
  for (int h = 0; h < HEADS; ++h) m[h] = -3.402823466e38f;
  for (int r = start + tid; r < end; r += 256) {
    const float* lp = wts + (size_t)r * HEADS;
#pragma unroll
    for (int h = 0; h < HEADS; ++h) m[h] = fmaxf(m[h], lp[h]);
  }
#pragma unroll
  for (int off = 16; off > 0; off >>= 1)
#pragma unroll
    for (int h = 0; h < HEADS; ++h) m[h] = fmaxf(m[h], __shfl_xor(m[h], off, 32));
  if (ln == 0)
#pragma unroll
    for (int h = 0; h < HEADS; ++h) warpred[wid][h] = m[h];
  __syncthreads();
  if (tid < HEADS) {
    float v = -3.402823466e38f;
#pragma unroll
    for (int w = 0; w < 8; ++w) v = fmaxf(v, warpred[w][tid]);
    if (v < -3.3e38f) v = 0.0f;            // empty-segment guard (isfinite)
    smax[tid] = v;
  }
  __syncthreads();

  // ---- pass 2: per-head sum of exp ----
  float s[HEADS];
#pragma unroll
  for (int h = 0; h < HEADS; ++h) s[h] = 0.0f;
  for (int r = start + tid; r < end; r += 256) {
    const float* lp = wts + (size_t)r * HEADS;
#pragma unroll
    for (int h = 0; h < HEADS; ++h) s[h] += __expf(lp[h] - smax[h]);
  }
#pragma unroll
  for (int off = 16; off > 0; off >>= 1)
#pragma unroll
    for (int h = 0; h < HEADS; ++h) s[h] += __shfl_xor(s[h], off, 32);
  if (ln == 0)
#pragma unroll
    for (int h = 0; h < HEADS; ++h) warpred[wid][h] = s[h];
  __syncthreads();
  if (tid < HEADS) {
    float v = 0.0f;
#pragma unroll
    for (int w = 0; w < 8; ++w) v += warpred[w][tid];
    if (!(v > 0.0f)) v = 1.0f;             // denom > 0 guard
    sinv[tid] = 1.0f / v;
  }
  __syncthreads();

  // ---- pass 3a: weights (in place) + entropy partial ----
  float ent = 0.0f;
  const int cnt8 = (end - start) * HEADS;
  for (int t = tid; t < cnt8; t += 256) {
    const int r = start + (t >> 3);
    const int h = t & 7;
    const size_t i = (size_t)r * HEADS + h;
    float w = __expf(wts[i] - smax[h]) * sinv[h];
    wts[i] = w;
    ent += w * __logf(w + EPS_W);
  }
#pragma unroll
  for (int off = 16; off > 0; off >>= 1) ent += __shfl_xor(ent, off, 32);
  if (ln == 0) warpred[wid][0] = ent;
  __threadfence_block();
  __syncthreads();
  if (tid == 0) {
    float e = 0.0f;
#pragma unroll
    for (int w = 0; w < 8; ++w) e += warpred[w][0];
    atomicAdd(out + ENT_OFF, -e * (1.0f / (float)(NGRAPH * HEADS)));
  }

  // ---- pass 3b: weighted scatter-sum -> graph_z (x read once, coalesced) ----
  const int c0 = tid, c1 = tid + 256;
  const int h0 = c0 >> 6, h1 = c1 >> 6;    // head = col / HEAD_DIM
  float a0 = 0.0f, a1 = 0.0f;
  for (int r = start; r < end; ++r) {
    const float* wr = wts + (size_t)r * HEADS;
    const float* xr = x + (size_t)r * HIDDEN;
    a0 = fmaf(xr[c0], wr[h0], a0);
    a1 = fmaf(xr[c1], wr[h1], a1);
  }
  gz[(size_t)g * HIDDEN + c0] = a0;
  gz[(size_t)g * HIDDEN + c1] = a1;
}

// ---------------------------------------------------------------------------
extern "C" void kernel_launch(void* const* d_in, const int* in_sizes, int n_in,
                              void* d_out, int out_size, void* d_ws, size_t ws_size,
                              hipStream_t stream) {
  (void)in_sizes; (void)n_in; (void)out_size; (void)ws_size;
  const float* x     = (const float*)d_in[0];
  const int*   batch = (const int*)d_in[1];
  const float* gamma = (const float*)d_in[2];
  const float* beta  = (const float*)d_in[3];
  const float* W1    = (const float*)d_in[4];
  const float* b1    = (const float*)d_in[5];
  const float* W2    = (const float*)d_in[6];
  const float* b2    = (const float*)d_in[7];
  float* out = (float*)d_out;

  _Float16* w1cm = (_Float16*)d_ws;                   // 256 KB
  _Float16* w2cm = w1cm + (size_t)HIDDEN * HALF_H;    // 8 KB

  pack_weights<<<256, 256, 0, stream>>>(W1, W2, w1cm, w2cm, out);
  ln_mlp_logits<<<NROWS / 32, 64, 0, stream>>>(x, gamma, beta, w1cm, b1,
                                               w2cm, b2, out + W_OFF);
  segment_softmax_pool<<<NGRAPH, 256, 0, stream>>>(x, batch, out);
}
